// GATPolicy_71494025609301
// MI455X (gfx1250) — matrix operand order repaired and checked by reading, not compiled, not run on hardware
//
#include <hip/hip_runtime.h>
#include <hip/hip_bf16.h>

#define N_NODES_C 100000
#define N_EDGES_C 1600000
#define E_TOT_C   (N_EDGES_C + N_NODES_C)
#define DIM       128
#define ACT       8
#define NGRAPH    256
#define NEG_SLOPE 0.2f

typedef __attribute__((ext_vector_type(16))) _Float16 v16h;
typedef __attribute__((ext_vector_type(8)))  float    v8f;

// ---------------------------------------------------------------------------
// H = X * W^T (X:[n,128] f32; W:[128,128] f32 row-major W[o][c]); also
// AS = H . att_src, AD = H . att_dst.
// 8 waves/block; each wave computes a 16-row x 128-col strip with WMMA.
// A/B converted to f16 in LDS; accumulate f32 (v_wmma_f32_16x16x32_f16).
// ---------------------------------------------------------------------------
__global__ __launch_bounds__(256)
void gat_gemm_attn(const float* __restrict__ X, const float* __restrict__ W,
                   const float* __restrict__ att_s, const float* __restrict__ att_d,
                   float* __restrict__ H, float* __restrict__ AS, float* __restrict__ AD,
                   int nrows)
{
    __shared__ _Float16 Wlds[DIM][DIM];   // row-major copy of W: Wlds[o][c] (B = W^T)
    __shared__ _Float16 Alds[8][16][DIM]; // per-wave 16x128 A tile in f16

    const int tid  = threadIdx.x;
    const int wv   = tid >> 5;
    const int lane = tid & 31;
    const int nl   = lane & 15;   // M (for A) / N (for B,C)
    const int half = lane >> 4;   // K-half selector

    // Stage W as f16 (contiguous global reads)
    for (int i = tid; i < DIM * DIM; i += 256)
        Wlds[i >> 7][i & 127] = (_Float16)W[i];

    // Stage this wave's 16 rows of X as f16
    const int row0 = blockIdx.x * 128 + wv * 16;
    for (int i = lane; i < 16 * DIM; i += 32) {
        int r = i >> 7, c = i & 127;
        float v = (row0 + r < nrows) ? X[(size_t)(row0 + r) * DIM + c] : 0.f;
        Alds[wv][r][c] = (_Float16)v;
    }
    __syncthreads();

    v8f zero = {};
    v8f acc[8];
#pragma unroll
    for (int t = 0; t < 8; ++t) acc[t] = zero;

#pragma unroll
    for (int kt = 0; kt < 4; ++kt) {
        // A fragment 16x32 f16: lane(nl,half) -> elems i<8: K=kt*32+half*8+i,
        //                                        i>=8:   K=kt*32+16+half*8+(i-8)
        v16h a;
        const _Float16* ap = &Alds[wv][nl][kt * 32 + half * 8];
#pragma unroll
        for (int i = 0; i < 8; ++i) { a[i] = ap[i]; a[8 + i] = ap[16 + i]; }

#pragma unroll
        for (int t = 0; t < 8; ++t) {
            // B fragment 32x16 f16: lane(nl,half) elem i<8 -> B[kt*32+half*8+i][t*16+nl], i>=8 -> B[kt*32+16+half*8+(i-8)][t*16+nl]
            v16h b;
            const _Float16* bp = &Wlds[t * 16 + nl][kt * 32 + half * 8];
#pragma unroll
            for (int i = 0; i < 8; ++i) { b[i] = bp[i]; b[8 + i] = bp[16 + i]; }
            acc[t] = __builtin_amdgcn_wmma_f32_16x16x32_f16(
                false, a, false, b, (short)0, acc[t], false, false);
        }
    }

    // Per-lane attention partials (column n = t*16 + nl)
    float asl[8], adl[8];
#pragma unroll
    for (int t = 0; t < 8; ++t) { asl[t] = att_s[t * 16 + nl]; adl[t] = att_d[t * 16 + nl]; }

#pragma unroll
    for (int r = 0; r < 8; ++r) {
        const int row = row0 + half * 8 + r;
        float s = 0.f, d = 0.f;
        if (row < nrows) {
            float* hp = H + (size_t)row * DIM + nl;
#pragma unroll
            for (int t = 0; t < 8; ++t) {
                float v = acc[t][r];
                hp[t * 16] = v;
                s += v * asl[t];
                d += v * adl[t];
            }
        }
        // reduce across the 16 lanes of this half (row uniform within half)
#pragma unroll
        for (int m = 8; m >= 1; m >>= 1) {
            s += __shfl_xor(s, m, 32);
            d += __shfl_xor(d, m, 32);
        }
        if (nl == 0 && row < nrows) { AS[row] = s; AD[row] = d; }
    }
}

// ---------------------------------------------------------------------------
// Edge phase: logits + leaky-relu + atomic segment max (ordered-uint keys)
// ---------------------------------------------------------------------------
__global__ __launch_bounds__(256)
void edge_logits_max(const int* __restrict__ ei, const float* __restrict__ AS,
                     const float* __restrict__ AD, float* __restrict__ ebuf,
                     unsigned* __restrict__ mkey)
{
    int i = blockIdx.x * blockDim.x + threadIdx.x;
    if (i >= E_TOT_C) return;
    int s, d;
    if (i < N_EDGES_C) { s = ei[i]; d = ei[N_EDGES_C + i]; }
    else               { s = d = i - N_EDGES_C; }   // self-loop
    float e = AS[s] + AD[d];
    e = (e > 0.f) ? e : NEG_SLOPE * e;
    ebuf[i] = e;
    unsigned b = __float_as_uint(e);
    unsigned key = (b & 0x80000000u) ? ~b : (b | 0x80000000u);
    atomicMax(&mkey[d], key);
}

__global__ __launch_bounds__(256)
void edge_exp_sum(const int* __restrict__ ei, const unsigned* __restrict__ mkey,
                  float* __restrict__ ebuf, float* __restrict__ denom)
{
    int i = blockIdx.x * blockDim.x + threadIdx.x;
    if (i >= E_TOT_C) return;
    int d = (i < N_EDGES_C) ? ei[N_EDGES_C + i] : (i - N_EDGES_C);
    unsigned key = mkey[d];
    unsigned mb = (key & 0x80000000u) ? (key & 0x7FFFFFFFu) : ~key;
    float ex = __expf(ebuf[i] - __uint_as_float(mb));
    ebuf[i] = ex;
    atomicAdd(&denom[d], ex);
}

// one wave per edge: 32 lanes x float4 = 128 features
__global__ __launch_bounds__(256)
void edge_aggregate(const int* __restrict__ ei, const float* __restrict__ ebuf,
                    const float* __restrict__ denom, const float* __restrict__ H,
                    float* __restrict__ agg)
{
    int w = (blockIdx.x * 256 + threadIdx.x) >> 5;
    int lane = threadIdx.x & 31;
    if (w >= E_TOT_C) return;
    int s, d;
    if (w < N_EDGES_C) { s = ei[w]; d = ei[N_EDGES_C + w]; }
    else               { s = d = w - N_EDGES_C; }
    float alpha = ebuf[w] / denom[d];
    const float4 h4 = ((const float4*)(H + (size_t)s * DIM))[lane];
    float* o = agg + (size_t)d * DIM + lane * 4;
    atomicAdd(o + 0, alpha * h4.x);
    atomicAdd(o + 1, alpha * h4.y);
    atomicAdd(o + 2, alpha * h4.z);
    atomicAdd(o + 3, alpha * h4.w);
}

__global__ __launch_bounds__(256)
void bias_relu(float* __restrict__ A, const float* __restrict__ b, size_t n)
{
    size_t i = (size_t)blockIdx.x * blockDim.x + threadIdx.x;
    if (i < n) {
        float v = A[i] + b[i & (DIM - 1)];
        A[i] = v > 0.f ? v : 0.f;
    }
}

// ---------------------------------------------------------------------------
// Head: warp per node, tanh(h @ Wl^T + bl), atomic mean-pool per graph
// ---------------------------------------------------------------------------
__global__ __launch_bounds__(256)
void head_pool(const float* __restrict__ H, const float* __restrict__ Wl,
               const float* __restrict__ bl, const int* __restrict__ batch,
               float* __restrict__ psum, float* __restrict__ pcnt, int n)
{
    int node = (blockIdx.x * 256 + threadIdx.x) >> 5;
    int lane = threadIdx.x & 31;
    if (node >= n) return;
    const float4 h4 = ((const float4*)(H + (size_t)node * DIM))[lane];
    int g = batch[node];
#pragma unroll
    for (int o = 0; o < ACT; ++o) {
        float4 w4 = ((const float4*)(Wl + o * DIM))[lane];
        float p = h4.x * w4.x + h4.y * w4.y + h4.z * w4.z + h4.w * w4.w;
#pragma unroll
        for (int m = 16; m >= 1; m >>= 1) p += __shfl_xor(p, m, 32);
        if (lane == 0) atomicAdd(&psum[g * ACT + o], tanhf(p + bl[o]));
    }
    if (lane == 0) atomicAdd(&pcnt[g], 1.0f);
}

__global__ void pool_finalize(const float* __restrict__ psum,
                              const float* __restrict__ pcnt, float* __restrict__ out)
{
    int i = blockIdx.x * blockDim.x + threadIdx.x;
    if (i < NGRAPH * ACT) out[i] = psum[i] / fmaxf(pcnt[i / ACT], 1.0f);
}

// ---------------------------------------------------------------------------
extern "C" void kernel_launch(void* const* d_in, const int* in_sizes, int n_in,
                              void* d_out, int out_size, void* d_ws, size_t ws_size,
                              hipStream_t stream)
{
    const float* x     = (const float*)d_in[0];
    const int*   ei    = (const int*)d_in[1];    // [2, E] int32
    const int*   batch = (const int*)d_in[2];
    const float* Wl    = (const float*)d_in[15];
    const float* bl    = (const float*)d_in[16];

    float* ws = (float*)d_ws;
    const size_t NF = (size_t)N_NODES_C * DIM;
    float*    bufA  = ws;
    float*    bufB  = bufA + NF;
    float*    hW    = bufB + NF;
    float*    AS    = hW + NF;
    float*    AD    = AS + N_NODES_C;
    float*    ebuf  = AD + N_NODES_C;
    unsigned* mkey  = (unsigned*)(ebuf + E_TOT_C);
    float*    denom = (float*)(mkey + N_NODES_C);
    float*    psum  = denom + N_NODES_C;
    float*    pcnt  = psum + NGRAPH * ACT;

    const int gemm_blocks = (N_NODES_C + 127) / 128;
    const int edge_blocks = (E_TOT_C + 255) / 256;
    const int eagg_blocks = (E_TOT_C + 7) / 8;
    const int br_blocks   = (int)((NF + 255) / 256);
    const int head_blocks = (N_NODES_C + 7) / 8;

    const float* cur = x;
    for (int layer = 0; layer < 3; ++layer) {
        const float* W   = (const float*)d_in[3 + 4 * layer];
        const float* as_ = (const float*)d_in[4 + 4 * layer];
        const float* ad_ = (const float*)d_in[5 + 4 * layer];
        const float* bb  = (const float*)d_in[6 + 4 * layer];
        float* agg = (layer & 1) ? bufB : bufA;   // 0->A, 1->B, 2->A

        gat_gemm_attn<<<gemm_blocks, 256, 0, stream>>>(cur, W, as_, ad_, hW, AS, AD, N_NODES_C);
        hipMemsetAsync(mkey, 0, N_NODES_C * sizeof(unsigned), stream);
        hipMemsetAsync(denom, 0, N_NODES_C * sizeof(float), stream);
        hipMemsetAsync(agg, 0, NF * sizeof(float), stream);
        edge_logits_max<<<edge_blocks, 256, 0, stream>>>(ei, AS, AD, ebuf, mkey);
        edge_exp_sum<<<edge_blocks, 256, 0, stream>>>(ei, mkey, ebuf, denom);
        edge_aggregate<<<eagg_blocks, 256, 0, stream>>>(ei, ebuf, denom, hW, agg);
        bias_relu<<<br_blocks, 256, 0, stream>>>(agg, bb, NF);
        cur = agg;
    }

    hipMemsetAsync(psum, 0, (NGRAPH * ACT + NGRAPH) * sizeof(float), stream);
    head_pool<<<head_blocks, 256, 0, stream>>>(cur, Wl, bl, batch, psum, pcnt, N_NODES_C);
    pool_finalize<<<(NGRAPH * ACT + 255) / 256, 256, 0, stream>>>(psum, pcnt, (float*)d_out);
}
